// Powerformer_backbone_13700945674564
// MI455X (gfx1250) — matrix operand order repaired and checked by reading, not compiled, not run on hardware
//
#include <hip/hip_runtime.h>

// ---------------------------------------------------------------------------
// Powerformer backbone for MI455X (gfx1250): bf16 WMMA + TDM tensor DMA.
// All WMMA operands are stored K-contiguous so fragment loads are wide
// (ds_load_b128 / global b128) instead of u16 gather soup.
// ---------------------------------------------------------------------------

typedef __bf16 bf16_t;
typedef __bf16 v16bf __attribute__((ext_vector_type(16)));
typedef __bf16 bfx8  __attribute__((ext_vector_type(8)));
typedef float  v8f   __attribute__((ext_vector_type(8)));
typedef unsigned int u32x4 __attribute__((ext_vector_type(4)));
typedef int          i32x8 __attribute__((ext_vector_type(8)));
typedef int          i32x4 __attribute__((ext_vector_type(4)));

#define WMMA_BF16(a, b, c) \
  __builtin_amdgcn_wmma_f32_16x16x32_bf16(false, (a), false, (b), (short)0, (c), false, false)

// ----- problem dims --------------------------------------------------------
constexpr int BS = 16, NV = 7, CTX = 1536;
constexpr int PLEN = 16, PSTRIDE = 8, PN = 192;
constexpr int D = 512, H = 8, DFF = 2048, L = 3, TW = 336, DK = 64;
constexpr int BATCH = BS * NV;          // 112
constexpr int ROWS  = BATCH * PN;       // 21504
constexpr int NF    = D * PN;           // 98304
constexpr float ATT_SCALE = 0.125f;     // DK^-0.5

// ----- element counts for workspace ---------------------------------------
constexpr size_t E_WQKV = (size_t)L * D * D;
constexpr size_t E_W1   = (size_t)L * D * DFF;
constexpr size_t E_W2   = (size_t)L * DFF * D;
constexpr size_t E_WH   = (size_t)NF * TW;
constexpr size_t E_ACT  = (size_t)ROWS * D;
constexpr size_t E_FF   = (size_t)ROWS * DFF;
constexpr size_t E_SC   = (size_t)BATCH * H * PN * PN;

// ===========================================================================
// LDS raw offset (workgroup-relative) for TDM descriptors
// ===========================================================================
typedef __attribute__((address_space(3))) void lds_void;
__device__ __forceinline__ unsigned lds_addr_of(void* p) {
  return (unsigned)(unsigned long long)(lds_void*)p;
}

// ===========================================================================
// WMMA fragment helpers (ISA §7.12.2), all K-contiguous, wide loads
// ===========================================================================
__device__ __forceinline__ v16bf frag_a(const bf16_t* p, int ld) {
  // A 16x32 tile, element (m,k) at p[m*ld + k]
  int lane = threadIdx.x & 31;
  const bf16_t* row = p + (lane & 15) * ld + ((lane >> 4) << 3);
  bfx8 lo = *(const bfx8*)(row);
  bfx8 hi = *(const bfx8*)(row + 16);
  return __builtin_shufflevector(lo, hi, 0, 1, 2, 3, 4, 5, 6, 7,
                                 8, 9, 10, 11, 12, 13, 14, 15);
}

__device__ __forceinline__ v16bf frag_bT(const bf16_t* p, int ld) {
  // B 32x16 tile with element (k,n) stored at p[n*ld + k]  (K contiguous)
  int lane = threadIdx.x & 31;
  const bf16_t* row = p + (lane & 15) * ld + ((lane >> 4) << 3);
  bfx8 lo = *(const bfx8*)(row);
  bfx8 hi = *(const bfx8*)(row + 16);
  return __builtin_shufflevector(lo, hi, 0, 1, 2, 3, 4, 5, 6, 7,
                                 8, 9, 10, 11, 12, 13, 14, 15);
}

// ===========================================================================
// Weight convert + transpose: out[l][n][k] = bf16(in[l][k][n])
// ===========================================================================
__global__ void f2bT_kernel(const float* __restrict__ in, bf16_t* __restrict__ out,
                            int Kd, int Nd, long long total) {
  long long i = (long long)blockIdx.x * blockDim.x + threadIdx.x;
  long long stride = (long long)gridDim.x * blockDim.x;
  long long pl = (long long)Kd * Nd;
  for (; i < total; i += stride) {
    long long l = i / pl, r = i % pl;
    int n = (int)(r / Kd), k = (int)(r % Kd);
    out[i] = (bf16_t)in[l * pl + (size_t)k * Nd + n];
  }
}

// ===========================================================================
// Patch embedding: src = patches @ W_P + b_P + W_pos
// ===========================================================================
__global__ __launch_bounds__(256) void patch_kernel(
    const float* __restrict__ z, const float* __restrict__ W_P,
    const float* __restrict__ b_P, const float* __restrict__ W_pos,
    float* __restrict__ srcF, bf16_t* __restrict__ srcB) {
  __shared__ float pv[PLEN];
  int row = blockIdx.x;                // bv*PN + p
  int bv = row / PN, p = row % PN;
  int tid = threadIdx.x;
  if (tid < PLEN) {
    int t = p * PSTRIDE + tid;
    pv[tid] = z[(size_t)bv * CTX + (t < CTX ? t : CTX - 1)];   // replication pad
  }
  __syncthreads();
#pragma unroll
  for (int i = 0; i < 2; ++i) {
    int d = tid + i * 256;
    float acc = b_P[d] + W_pos[(size_t)p * D + d];
#pragma unroll
    for (int j = 0; j < PLEN; ++j) acc += pv[j] * W_P[(size_t)j * D + d];
    srcF[(size_t)row * D + d] = acc;
    srcB[(size_t)row * D + d] = (bf16_t)acc;
  }
}

// ===========================================================================
// Generic bf16 GEMM: C[M,N] = A[M,K] @ BT[N,K]^T + bias
//   A tile (128x32) arrives via TDM tensor_load_to_lds (wave0 issues, double
//   buffered, TENSORcnt-synchronized).  B tile staged K-contiguous.
// mode 0: f32 out; 1: bf16 out; 2: bf16 out + exact GELU
// ===========================================================================
constexpr int GBM = 128, GBN = 128, GBK = 32;
constexpr int LDA2 = GBK;        // A tile rows contiguous (TDM writes packed)
constexpr int LDBT = GBK + 8;    // 40 elems -> 80B rows (16B aligned)

__global__ __launch_bounds__(256) void gemm_kernel(
    const bf16_t* __restrict__ A, const bf16_t* __restrict__ BT,
    const float* __restrict__ bias, float* __restrict__ Cf,
    bf16_t* __restrict__ Cb, int M, int N, int K, int mode) {
  __shared__ bf16_t As[2][GBM * LDA2];
  __shared__ bf16_t Bs[2][GBN * LDBT];
  int tid = threadIdx.x;
  int bm = blockIdx.y * GBM, bn = blockIdx.x * GBN;
  int wid = tid >> 5;
  int mw = wid >> 1, nw = wid & 1;

  auto tdm_issue = [&](int buf, int kt) {
    unsigned lds = lds_addr_of(&As[buf][0]);
    unsigned long long ga =
        (unsigned long long)(A + (size_t)bm * K + (size_t)kt * GBK);
    u32x4 g0;
    g0[0] = 1u;                                   // count=1, user descriptor
    g0[1] = lds;                                  // LDS byte address
    g0[2] = (unsigned)ga;                         // global addr [31:0]
    g0[3] = (unsigned)((ga >> 32) & 0x1FFFFFFu) | (2u << 30);  // addr hi | type=2
    unsigned d0 = (unsigned)K, d1 = (unsigned)M;
    unsigned t0 = (unsigned)GBK, t1 = (unsigned)GBM;
    unsigned long long s0 = (unsigned long long)K;  // dim0 stride (elements)
    i32x8 g1;
    g1[0] = 0x00010000;                           // data_size=1 (2 bytes)
    g1[1] = (int)(d0 << 16);                      // abar=0 | dim0 lo
    g1[2] = (int)((d0 >> 16) | ((d1 & 0xFFFFu) << 16));
    g1[3] = (int)((d1 >> 16) | (t0 << 16));       // dim1 hi | tile0
    g1[4] = (int)(t1 & 0xFFFFu);                  // tile1 | tile2=0
    g1[5] = (int)(unsigned)s0;                    // dim0 stride lo32
    g1[6] = (int)(unsigned)((s0 >> 32) & 0xFFFFu);
    g1[7] = 0;
    i32x4 g2 = {0, 0, 0, 0}, g3 = {0, 0, 0, 0};
    i32x8 g4 = {0, 0, 0, 0, 0, 0, 0, 0};
    __builtin_amdgcn_tensor_load_to_lds(g0, g1, g2, g3, g4, 0);
  };

  bfx8 rb[2];
  auto gloadB = [&](int kt) {
#pragma unroll
    for (int c = 0; c < 2; ++c) {
      int ch = tid + c * 256;
      int n = ch >> 2, kc = (ch & 3) << 3;
      rb[c] = *(const bfx8*)(BT + (size_t)(bn + n) * K + kt * GBK + kc);
    }
  };
  auto sstoreB = [&](int buf) {
#pragma unroll
    for (int c = 0; c < 2; ++c) {
      int ch = tid + c * 256;
      int n = ch >> 2, kc = (ch & 3) << 3;
      *(bfx8*)&Bs[buf][n * LDBT + kc] = rb[c];
    }
  };

  v8f acc[2][4];
#pragma unroll
  for (int i = 0; i < 2; ++i)
#pragma unroll
    for (int j = 0; j < 4; ++j) acc[i][j] = (v8f){0.f,0.f,0.f,0.f,0.f,0.f,0.f,0.f};

  int NT = K / GBK;
  if (wid == 0) tdm_issue(0, 0);
  gloadB(0);
  sstoreB(0);
  if (wid == 0) __builtin_amdgcn_s_wait_tensorcnt(0);
  __syncthreads();

  for (int kt = 0; kt < NT; ++kt) {
    int buf = kt & 1;
    if (kt + 1 < NT) {
      if (wid == 0) tdm_issue(buf ^ 1, kt + 1);
      gloadB(kt + 1);
    }
    v16bf af[2], bf_[4];
#pragma unroll
    for (int i = 0; i < 2; ++i)
      af[i] = frag_a(&As[buf][(mw * 32 + i * 16) * LDA2], LDA2);
#pragma unroll
    for (int j = 0; j < 4; ++j)
      bf_[j] = frag_bT(&Bs[buf][(nw * 64 + j * 16) * LDBT], LDBT);
#pragma unroll
    for (int i = 0; i < 2; ++i)
#pragma unroll
      for (int j = 0; j < 4; ++j) acc[i][j] = WMMA_BF16(af[i], bf_[j], acc[i][j]);
    if (kt + 1 < NT) {
      sstoreB(buf ^ 1);
      if (wid == 0) __builtin_amdgcn_s_wait_tensorcnt(0);
    }
    __syncthreads();
  }

  int lane = tid & 31;
  int nl = lane & 15, mb8 = (lane >> 4) << 3;
#pragma unroll
  for (int i = 0; i < 2; ++i)
#pragma unroll
    for (int j = 0; j < 4; ++j) {
      int col = bn + nw * 64 + j * 16 + nl;
      float bv = bias ? bias[col] : 0.f;
#pragma unroll
      for (int r = 0; r < 8; ++r) {
        int row = bm + mw * 32 + i * 16 + mb8 + r;
        float v = acc[i][j][r] + bv;
        size_t idx = (size_t)row * N + col;
        if (mode == 0) {
          Cf[idx] = v;
        } else {
          if (mode == 2) v = 0.5f * v * (1.f + erff(v * 0.70710678118654752f));
          Cb[idx] = (bf16_t)v;
        }
      }
    }
}

// ===========================================================================
// scores = q @ k^T * SCALE + prev + decay_mask   (per (b,h), 64 q-rows/block)
// k tile already K(=dk)-contiguous: frag_bT directly.
// ===========================================================================
__global__ __launch_bounds__(256) void scores_kernel(
    const bf16_t* __restrict__ qB, const bf16_t* __restrict__ kB,
    float* __restrict__ scF, const float* __restrict__ dm) {
  __shared__ bf16_t qs[64 * 72];
  __shared__ bf16_t ks[PN * 72];
  int tid = threadIdx.x;
  int bh = blockIdx.y, qt = blockIdx.x;
  int b = bh >> 3, h = bh & 7;
  int q0 = qt * 64;
#pragma unroll
  for (int c = 0; c < 2; ++c) {
    int ch = tid + c * 256;
    int r = ch >> 3, cc = (ch & 7) << 3;
    *(bfx8*)&qs[r * 72 + cc] =
        *(const bfx8*)(qB + (size_t)(b * PN + q0 + r) * D + h * DK + cc);
  }
#pragma unroll
  for (int c = 0; c < 6; ++c) {
    int ch = tid + c * 256;
    int r = ch >> 3, cc = (ch & 7) << 3;
    *(bfx8*)&ks[r * 72 + cc] =
        *(const bfx8*)(kB + (size_t)(b * PN + r) * D + h * DK + cc);
  }
  __syncthreads();

  int wid = tid >> 5;
  int mw = wid >> 1, nh = wid & 1;
  v8f acc[6];
#pragma unroll
  for (int j = 0; j < 6; ++j) acc[j] = (v8f){0.f,0.f,0.f,0.f,0.f,0.f,0.f,0.f};

#pragma unroll
  for (int kk = 0; kk < 2; ++kk) {
    v16bf a = frag_a(&qs[(mw * 16) * 72 + kk * 32], 72);
#pragma unroll
    for (int j = 0; j < 6; ++j) {
      v16bf bb = frag_bT(&ks[(nh * 96 + j * 16) * 72 + kk * 32], 72);
      acc[j] = WMMA_BF16(a, bb, acc[j]);
    }
  }

  int lane = tid & 31;
  int nl = lane & 15, mb8 = (lane >> 4) << 3;
#pragma unroll
  for (int j = 0; j < 6; ++j)
#pragma unroll
    for (int r = 0; r < 8; ++r) {
      int qi = q0 + mw * 16 + mb8 + r;
      int kj = nh * 96 + j * 16 + nl;
      size_t idx = ((size_t)bh * PN + qi) * PN + kj;
      scF[idx] = acc[j][r] * ATT_SCALE + scF[idx] + dm[qi * PN + kj];
    }
}

// ===========================================================================
// Row softmax over scores (192 wide), write bf16 attn
// ===========================================================================
__global__ __launch_bounds__(192) void softmax_kernel(
    const float* __restrict__ scF, bf16_t* __restrict__ attnB) {
  __shared__ float red[256];
  size_t row = blockIdx.x;
  int tid = threadIdx.x;
  float v = scF[row * PN + tid];
  red[tid] = v;
  if (tid < 64) red[192 + tid] = -__builtin_inff();
  __syncthreads();
  for (int s = 128; s > 0; s >>= 1) {
    if (tid < s) red[tid] = fmaxf(red[tid], red[tid + s]);
    __syncthreads();
  }
  float mx = red[0];
  __syncthreads();
  float e = expf(v - mx);
  red[tid] = e;
  if (tid < 64) red[192 + tid] = 0.f;
  __syncthreads();
  for (int s = 128; s > 0; s >>= 1) {
    if (tid < s) red[tid] += red[tid + s];
    __syncthreads();
  }
  attnB[row * PN + tid] = (bf16_t)(e / red[0]);
}

// ===========================================================================
// o = attn @ v  (per (b,h)).  v tile transposed in-kernel to K-contiguous
// (coalesced u16 gathers -> packed b128 LDS stores), frags all wide.
// ===========================================================================
__global__ __launch_bounds__(256) void attnv_kernel(
    const bf16_t* __restrict__ attnB, const bf16_t* __restrict__ vB,
    bf16_t* __restrict__ oB) {
  __shared__ bf16_t as_[64 * 200];
  __shared__ bf16_t vsT[64 * 200];   // [dk][key] : K(=key)-contiguous
  int tid = threadIdx.x;
  int bh = blockIdx.y, qt = blockIdx.x;
  int b = bh >> 3, h = bh & 7;
  int q0 = qt * 64;
#pragma unroll
  for (int c = 0; c < 6; ++c) {
    int ch = tid + c * 256;
    int r = ch / 24, cc = (ch % 24) << 3;
    *(bfx8*)&as_[r * 200 + cc] =
        *(const bfx8*)(attnB + ((size_t)bh * PN + q0 + r) * PN + cc);
  }
#pragma unroll
  for (int c = 0; c < 6; ++c) {
    int ch = tid + c * 256;
    int dk = ch & 63, kc = ch >> 6;           // 64 dk x 24 key-chunks
    bfx8 vv;
#pragma unroll
    for (int j = 0; j < 8; ++j)
      vv[j] = vB[(size_t)(b * PN + kc * 8 + j) * D + h * DK + dk];
    *(bfx8*)&vsT[dk * 200 + kc * 8] = vv;
  }
  __syncthreads();

  int wid = tid >> 5;
  int mw = wid & 3, nw = wid >> 2;
  v8f acc[2];
  acc[0] = (v8f){0.f,0.f,0.f,0.f,0.f,0.f,0.f,0.f};
  acc[1] = (v8f){0.f,0.f,0.f,0.f,0.f,0.f,0.f,0.f};
#pragma unroll
  for (int kt = 0; kt < 6; ++kt) {
    v16bf a = frag_a(&as_[(mw * 16) * 200 + kt * 32], 200);
#pragma unroll
    for (int j = 0; j < 2; ++j) {
      v16bf bb = frag_bT(&vsT[(nw * 32 + j * 16) * 200 + kt * 32], 200);
      acc[j] = WMMA_BF16(a, bb, acc[j]);
    }
  }
  int lane = tid & 31;
  int nl = lane & 15, mb8 = (lane >> 4) << 3;
#pragma unroll
  for (int j = 0; j < 2; ++j)
#pragma unroll
    for (int r = 0; r < 8; ++r) {
      int qi = q0 + mw * 16 + mb8 + r;
      int col = h * DK + nw * 32 + j * 16 + nl;
      oB[(size_t)(b * PN + qi) * D + col] = (bf16_t)acc[j][r];
    }
}

// ===========================================================================
// srcF = LN(srcF + addF); also write bf16 copy.  One row (512) per block.
// ===========================================================================
__global__ __launch_bounds__(128) void addln_kernel(
    float* __restrict__ srcF, const float* __restrict__ addF,
    const float* __restrict__ g, const float* __restrict__ be,
    bf16_t* __restrict__ srcB) {
  __shared__ float rs[128], rq[128];
  size_t row = blockIdx.x;
  int tid = threadIdx.x;
  float x[4];
  float s = 0.f, q = 0.f;
#pragma unroll
  for (int i = 0; i < 4; ++i) {
    int c = tid + i * 128;
    x[i] = srcF[row * D + c] + addF[row * D + c];
    s += x[i];
    q += x[i] * x[i];
  }
  rs[tid] = s;
  rq[tid] = q;
  __syncthreads();
  for (int st = 64; st > 0; st >>= 1) {
    if (tid < st) { rs[tid] += rs[tid + st]; rq[tid] += rq[tid + st]; }
    __syncthreads();
  }
  float mu = rs[0] * (1.f / D);
  float var = rq[0] * (1.f / D) - mu * mu;
  float inv = rsqrtf(var + 1e-5f);
#pragma unroll
  for (int i = 0; i < 4; ++i) {
    int c = tid + i * 128;
    float y = g[c] * (x[i] - mu) * inv + be[c];
    srcF[row * D + c] = y;
    srcB[row * D + c] = (bf16_t)y;
  }
}

// ===========================================================================
// srcT[bv][d][p] = bf16(srcF[bv][p][d])   (flatten-head transpose)
// ===========================================================================
__global__ void transpose_kernel(const float* __restrict__ srcF,
                                 bf16_t* __restrict__ srcT) {
  size_t i = (size_t)blockIdx.x * blockDim.x + threadIdx.x;
  if (i >= E_ACT) return;
  int p = (int)(i % PN);
  size_t r = i / PN;
  int d = (int)(r % D);
  int bv = (int)(r / D);
  srcT[i] = (bf16_t)srcF[((size_t)bv * PN + p) * D + d];
}

// ===========================================================================
// Head: out[112,336] = srcT[112,98304] @ W_head + b_head
// W_head pre-transposed -> both operands are contiguous 16B global loads.
// grid (21 ntiles, 7 mtiles), 4 waves split K, WMMA + LDS reduction.
// ===========================================================================
__global__ __launch_bounds__(128) void head_kernel(
    const bf16_t* __restrict__ srcT, const bf16_t* __restrict__ WhT,
    const float* __restrict__ bh_, float* __restrict__ out) {
  __shared__ float part[4][256];
  int tid = threadIdx.x;
  int wid = tid >> 5, lane = tid & 31;
  int nt = blockIdx.x, mt = blockIdx.y;
  const int KW = NF / 4;   // 24576 per wave
  int k0 = wid * KW;

  v8f acc = (v8f){0.f,0.f,0.f,0.f,0.f,0.f,0.f,0.f};
  int ml = lane & 15;
  int kb = (lane >> 4) << 3;     // A half offset: 0 or 8
  int kh = (lane >> 4) << 3;     // B half offset: 0 or 8
  int nl = lane & 15;
  const bf16_t* arow = srcT + (size_t)(mt * 16 + ml) * NF;
  const bf16_t* brow = WhT + (size_t)(nt * 16 + nl) * NF;

  for (int kk = k0; kk < k0 + KW; kk += 32) {
    __builtin_prefetch(arow + kk + 256, 0, 0);
    __builtin_prefetch(brow + kk + 256, 0, 0);
    bfx8 alo = *(const bfx8*)(arow + kk + kb);
    bfx8 ahi = *(const bfx8*)(arow + kk + kb + 16);
    v16bf a = __builtin_shufflevector(alo, ahi, 0, 1, 2, 3, 4, 5, 6, 7,
                                      8, 9, 10, 11, 12, 13, 14, 15);
    bfx8 blo = *(const bfx8*)(brow + kk + kh);
    bfx8 bhi = *(const bfx8*)(brow + kk + kh + 16);
    v16bf b = __builtin_shufflevector(blo, bhi, 0, 1, 2, 3, 4, 5, 6, 7,
                                      8, 9, 10, 11, 12, 13, 14, 15);
    acc = WMMA_BF16(a, b, acc);
  }

  int mb8 = (lane >> 4) << 3;
#pragma unroll
  for (int r = 0; r < 8; ++r) part[wid][(mb8 + r) * 16 + nl] = acc[r];
  __syncthreads();
  for (int e = tid; e < 256; e += 128) {
    float s = part[0][e] + part[1][e] + part[2][e] + part[3][e];
    int mloc = e >> 4, nloc = e & 15;
    out[(size_t)(mt * 16 + mloc) * TW + nt * 16 + nloc] = s + bh_[nt * 16 + nloc];
  }
}

// ===========================================================================
// Host launcher
// ===========================================================================
extern "C" void kernel_launch(void* const* d_in, const int* in_sizes, int n_in,
                              void* d_out, int out_size, void* d_ws, size_t ws_size,
                              hipStream_t stream) {
  (void)in_sizes; (void)n_in; (void)out_size; (void)ws_size;

  const float* z      = (const float*)d_in[0];
  const float* W_P    = (const float*)d_in[1];
  const float* b_P    = (const float*)d_in[2];
  const float* W_pos  = (const float*)d_in[3];
  const float* Wq     = (const float*)d_in[4];
  const float* bq     = (const float*)d_in[5];
  const float* Wk     = (const float*)d_in[6];
  const float* bk     = (const float*)d_in[7];
  const float* Wv     = (const float*)d_in[8];
  const float* bv     = (const float*)d_in[9];
  const float* Wo     = (const float*)d_in[10];
  const float* bo     = (const float*)d_in[11];
  const float* ln1_g  = (const float*)d_in[12];
  const float* ln1_b  = (const float*)d_in[13];
  const float* W1     = (const float*)d_in[14];
  const float* b1     = (const float*)d_in[15];
  const float* W2     = (const float*)d_in[16];
  const float* b2     = (const float*)d_in[17];
  const float* ln2_g  = (const float*)d_in[18];
  const float* ln2_b  = (const float*)d_in[19];
  const float* W_head = (const float*)d_in[20];
  const float* b_head = (const float*)d_in[21];
  const float* dmask  = (const float*)d_in[22];

  char* w = (char*)d_ws;
  size_t off = 0;
  auto take = [&](size_t bytes) {
    size_t r = off;
    off += (bytes + 255) & ~(size_t)255;
    return r;
  };

  bf16_t* WqT = (bf16_t*)(w + take(E_WQKV * 2));   // [L][N][K] transposed bf16
  bf16_t* WkT = (bf16_t*)(w + take(E_WQKV * 2));
  bf16_t* WvT = (bf16_t*)(w + take(E_WQKV * 2));
  bf16_t* WoT = (bf16_t*)(w + take(E_WQKV * 2));
  bf16_t* W1T = (bf16_t*)(w + take(E_W1 * 2));
  bf16_t* W2T = (bf16_t*)(w + take(E_W2 * 2));
  bf16_t* WHT = (bf16_t*)(w + take(E_WH * 2));     // [TW][NF]
  float*  srcF  = (float*) (w + take(E_ACT * 4));
  bf16_t* srcB  = (bf16_t*)(w + take(E_ACT * 2));
  bf16_t* qB    = (bf16_t*)(w + take(E_ACT * 2));
  bf16_t* kB    = (bf16_t*)(w + take(E_ACT * 2));
  bf16_t* vB    = (bf16_t*)(w + take(E_ACT * 2));
  bf16_t* oB    = (bf16_t*)(w + take(E_ACT * 2));
  float*  scF   = (float*) (w + take(E_SC * 4));
  float*  tmpF  = (float*) (w + take(E_ACT * 4));
  char*   shreg = w + take(E_FF * 2);   // attn (66MB) / ffn (88MB) share
  bf16_t* attnB = (bf16_t*)shreg;
  bf16_t* ffB   = (bf16_t*)shreg;
  bf16_t* srcT  = qB;                   // q dead after attention

  auto convT = [&](const float* src, bf16_t* dst, int Kd, int Nd, size_t n) {
    size_t g = (n + 255) / 256;
    if (g > 4096) g = 4096;
    f2bT_kernel<<<dim3((unsigned)g), dim3(256), 0, stream>>>(
        src, dst, Kd, Nd, (long long)n);
  };
  convT(Wq, WqT, D, D, E_WQKV);
  convT(Wk, WkT, D, D, E_WQKV);
  convT(Wv, WvT, D, D, E_WQKV);
  convT(Wo, WoT, D, D, E_WQKV);
  convT(W1, W1T, D, DFF, E_W1);
  convT(W2, W2T, DFF, D, E_W2);
  convT(W_head, WHT, NF, TW, E_WH);

  patch_kernel<<<dim3(ROWS), dim3(256), 0, stream>>>(z, W_P, b_P, W_pos, srcF, srcB);
  (void)hipMemsetAsync(scF, 0, E_SC * 4, stream);   // residual attention carry = 0

  auto gemm = [&](const bf16_t* Aa, const bf16_t* Bt, const float* bias,
                  float* Cf, bf16_t* Cb, int M, int N, int K, int mode) {
    gemm_kernel<<<dim3(N / GBN, M / GBM), dim3(256), 0, stream>>>(
        Aa, Bt, bias, Cf, Cb, M, N, K, mode);
  };

  for (int l = 0; l < L; ++l) {
    const bf16_t* Wq_l = WqT + (size_t)l * D * D;
    const bf16_t* Wk_l = WkT + (size_t)l * D * D;
    const bf16_t* Wv_l = WvT + (size_t)l * D * D;
    const bf16_t* Wo_l = WoT + (size_t)l * D * D;
    const bf16_t* W1_l = W1T + (size_t)l * D * DFF;
    const bf16_t* W2_l = W2T + (size_t)l * DFF * D;

    gemm(srcB, Wq_l, bq + (size_t)l * D, nullptr, qB, ROWS, D, D, 1);
    gemm(srcB, Wk_l, bk + (size_t)l * D, nullptr, kB, ROWS, D, D, 1);
    gemm(srcB, Wv_l, bv + (size_t)l * D, nullptr, vB, ROWS, D, D, 1);

    scores_kernel<<<dim3(3, BATCH * H), dim3(256), 0, stream>>>(
        qB, kB, scF, dmask + (size_t)l * PN * PN);
    softmax_kernel<<<dim3(BATCH * H * PN), dim3(192), 0, stream>>>(scF, attnB);
    attnv_kernel<<<dim3(3, BATCH * H), dim3(256), 0, stream>>>(attnB, vB, oB);

    gemm(oB, Wo_l, bo + (size_t)l * D, tmpF, nullptr, ROWS, D, D, 0);
    addln_kernel<<<dim3(ROWS), dim3(128), 0, stream>>>(
        srcF, tmpF, ln1_g + (size_t)l * D, ln1_b + (size_t)l * D, srcB);

    gemm(srcB, W1_l, b1 + (size_t)l * DFF, nullptr, ffB, ROWS, DFF, D, 2);
    gemm(ffB, W2_l, b2 + (size_t)l * D, tmpF, nullptr, ROWS, D, DFF, 0);
    addln_kernel<<<dim3(ROWS), dim3(128), 0, stream>>>(
        srcF, tmpF, ln2_g + (size_t)l * D, ln2_b + (size_t)l * D, srcB);
  }

  transpose_kernel<<<dim3((unsigned)((E_ACT + 255) / 256)), dim3(256), 0, stream>>>(
      srcF, srcT);
  head_kernel<<<dim3(TW / 16, BATCH / 16), dim3(128), 0, stream>>>(
      srcT, WHT, b_head, (float*)d_out);
}
